// GPT_90812788507197
// MI455X (gfx1250) — compile-verified, matrix operand order checked by reading, not run on hardware
//
#include <hip/hip_runtime.h>
#include <hip/hip_bf16.h>

typedef __attribute__((ext_vector_type(16))) __bf16 v16bf;
typedef __attribute__((ext_vector_type(8)))  float  v8f;
typedef unsigned short u16t;
typedef unsigned int   u32t;
typedef unsigned long long u64t;

#define D_MODEL 512
#define NLAYER  6
#define NHEAD   8
#define DHEAD   64
#define SEQ     2048
#define BATCH   2
#define ROWS    (BATCH*SEQ)   /* 4096 */
#define FFDIM   2048
#define VOCAB   32000
#define TD      (3*D_MODEL)   /* 1536 */

__device__ __forceinline__ u16t f2b(float f) {
  u32t u = __float_as_uint(f);
  u32t r = u + 0x7FFFu + ((u >> 16) & 1u);   // round-to-nearest-even bf16
  return (u16t)(r >> 16);
}

union FragU { v16bf v; u32t u[8]; };

// Load a 16x32 bf16 WMMA fragment (A layout, or B layout from a [col][k] image).
// Lane mapping per ISA: lanes 0-15 hold K = k0+0..7 and k0+16..23,
// lanes 16-31 hold K = k0+8..15 and k0+24..31 (pairs packed per dword).
__device__ __forceinline__ v16bf load_frag(const u16t* base, int row, int k0, int rs) {
  int lane = threadIdx.x & 31;
  const u16t* p = base + (size_t)(row + (lane & 15)) * rs + k0 + ((lane >> 4) << 3);
  FragU f;
  const u32t* p0 = (const u32t*)p;
  const u32t* p1 = (const u32t*)(p + 16);
  f.u[0] = p0[0]; f.u[1] = p0[1]; f.u[2] = p0[2]; f.u[3] = p0[3];
  f.u[4] = p1[0]; f.u[5] = p1[1]; f.u[6] = p1[2]; f.u[7] = p1[3];
  return f.v;
}

__device__ __forceinline__ v8f wmma_bf16(v16bf a, v16bf b, v8f c) {
  return __builtin_amdgcn_wmma_f32_16x16x32_bf16(false, a, false, b, (short)0, c, false, false);
}

// gfx1250 async global->LDS copy, 16 bytes per lane (ASYNCcnt-tracked, no VGPR data).
__device__ __forceinline__ void async_copy_b128(unsigned lds_off, const void* gaddr) {
  u64t ga = (u64t)(uintptr_t)gaddr;
  asm volatile("global_load_async_to_lds_b128 %0, %1, off"
               :: "v"(lds_off), "v"(ga) : "memory");
}
__device__ __forceinline__ void async_wait0() {
  asm volatile("s_wait_asynccnt 0x0" ::: "memory");
}
__device__ __forceinline__ void async_wait6() {   // allow newest 6 per-wave issues in flight
  asm volatile("s_wait_asynccnt 0x6" ::: "memory");
}
__device__ __forceinline__ unsigned lds_offset(const void* p) {
  return (unsigned)(uintptr_t)p;   // generic LDS ptr: low 32 bits = LDS byte offset
}

// ---------------------------------------------------------------------------
// Transposing fp32 -> bf16 weight conversion: in [K,N] fp32 -> out [N,K] bf16.
// grid (N/32, K/32, layers), block 256. LDS-tiled for coalescing both sides.
// ---------------------------------------------------------------------------
__global__ __launch_bounds__(256) void cvt_t_kernel(const float* __restrict__ in,
    u16t* __restrict__ out, int K, int N) {
  __shared__ u16t tile[32][33];
  size_t lbase = (size_t)blockIdx.z * K * N;
  int n0 = blockIdx.x * 32, k0 = blockIdx.y * 32;
  int tx = threadIdx.x & 31, ty = threadIdx.x >> 5;
  for (int j = ty; j < 32; j += 8)
    tile[j][tx] = f2b(in[lbase + (size_t)(k0 + j) * N + n0 + tx]);
  __syncthreads();
  for (int j = ty; j < 32; j += 8)
    out[lbase + (size_t)(n0 + j) * K + k0 + tx] = tile[tx][j];
}

// ---------------------------------------------------------------------------
// Per-layer V transpose: qkv bf16 [4096,1536] -> vt bf16 [B*H][DHEAD][SEQ]
// ---------------------------------------------------------------------------
__global__ __launch_bounds__(256) void vtrans_kernel(const u16t* __restrict__ qkv,
                                                     u16t* __restrict__ vt) {
  __shared__ u16t tile[32][33];
  int bh = blockIdx.z; int b = bh >> 3, h = bh & 7;
  int s0 = blockIdx.x * 32, d0 = blockIdx.y * 32;
  int tx = threadIdx.x & 31, ty = threadIdx.x >> 5;
  for (int j = ty; j < 32; j += 8)
    tile[j][tx] = qkv[(size_t)(b * SEQ + s0 + j) * TD + 1024 + h * DHEAD + d0 + tx];
  __syncthreads();
  for (int j = ty; j < 32; j += 8)
    vt[((size_t)bh * DHEAD + d0 + j) * SEQ + s0 + tx] = tile[tx][j];
}

// ---------------------------------------------------------------------------
// Embedding + sinusoidal PE  -> h fp32 [4096,512]
// ---------------------------------------------------------------------------
__global__ __launch_bounds__(512) void embed_kernel(const int* __restrict__ x,
    const float* __restrict__ emb, float* __restrict__ h) {
  int row = blockIdx.x;
  int d = threadIdx.x;
  int s = row & (SEQ - 1);
  int tok = x[row];
  int i2 = d & ~1;
  float div = __expf(-(float)i2 * (9.210340371976184f / 512.0f)); // ln(1e4)/D
  float ang = (float)s * div;
  float pe = (d & 1) ? __cosf(ang) : __sinf(ang);
  h[(size_t)row * D_MODEL + d] = emb[(size_t)tok * D_MODEL + d] + pe;
}

// ---------------------------------------------------------------------------
// LayerNorm: fp32 [4096,512] -> bf16 [4096,512].  One wave per row.
// ---------------------------------------------------------------------------
__global__ __launch_bounds__(256) void ln_kernel(const float* __restrict__ x,
    const float* __restrict__ g, const float* __restrict__ bta,
    u16t* __restrict__ out) {
  int lane = threadIdx.x & 31;
  int wid = threadIdx.x >> 5;
  int row = blockIdx.x * 8 + wid;
  const float* xr = x + (size_t)row * D_MODEL;
  float v[16];
  float s = 0.f, sq = 0.f;
  for (int j = 0; j < 4; j++) {
    float4 t = *(const float4*)&xr[lane * 16 + j * 4];
    v[j*4+0] = t.x; v[j*4+1] = t.y; v[j*4+2] = t.z; v[j*4+3] = t.w;
    s  += t.x + t.y + t.z + t.w;
    sq += t.x*t.x + t.y*t.y + t.z*t.z + t.w*t.w;
  }
  for (int off = 16; off >= 1; off >>= 1) {
    s  += __shfl_xor(s,  off, 32);
    sq += __shfl_xor(sq, off, 32);
  }
  float mean = s * (1.0f / 512.0f);
  float var  = sq * (1.0f / 512.0f) - mean * mean;
  float rstd = rsqrtf(var + 1e-5f);
  u16t* orow = out + (size_t)row * D_MODEL;
  for (int j = 0; j < 16; j++) {
    int c = lane * 16 + j;
    orow[c] = f2b((v[j] - mean) * rstd * g[c] + bta[c]);
  }
}

// ---------------------------------------------------------------------------
// Generic WMMA GEMM:  C[M,N] = A_bf16[M,K] @ Wt_bf16[N,K]^T + bias (+resid)
// block = 256 threads = 8 waves (4x2); block tile 128x64, BK=64.
// Double-buffered LDS; both A and Wt tiles staged with async global->LDS
// copies overlapped with WMMA compute of the previous tile.
// ---------------------------------------------------------------------------
__global__ __launch_bounds__(256) void gemm_kernel(
    const u16t* __restrict__ A, const u16t* __restrict__ Wt,
    const float* __restrict__ bias, const float* __restrict__ resid,
    float* __restrict__ outF, u16t* __restrict__ outB,
    int M, int N, int K, int relu) {
  __shared__ u16t sA[2][128 * 64];   // [m][k]  2 x 16KB
  __shared__ u16t sW[2][64 * 64];    // [n][k]  2 x 8KB
  int tid = threadIdx.x;
  int lane = tid & 31;
  int wid = tid >> 5;
  int wm = wid >> 1, wn = wid & 1;
  int m0 = blockIdx.y * 128;
  int n0 = blockIdx.x * 64;
  unsigned aoff[2] = { lds_offset(sA[0]), lds_offset(sA[1]) };
  unsigned woff[2] = { lds_offset(sW[0]), lds_offset(sW[1]) };

  // stage K-tile t into LDS buffer `buf` (6 async issues per thread/wave)
  auto stage = [&](int k0, int buf) {
    for (int r = 0; r < 4; r++) {               // A: 128 rows x 128B
      int lin = (tid + (r << 8)) << 4;
      int m = lin >> 7, kb = lin & 127;
      async_copy_b128(aoff[buf] + (unsigned)lin,
                      (const char*)(A + (size_t)(m0 + m) * K + k0) + kb);
    }
    for (int r = 0; r < 2; r++) {               // Wt: 64 rows x 128B
      int lin = (tid + (r << 8)) << 4;
      int n = lin >> 7, kb = lin & 127;
      async_copy_b128(woff[buf] + (unsigned)lin,
                      (const char*)(Wt + (size_t)(n0 + n) * K + k0) + kb);
    }
  };

  v8f z = {0.f,0.f,0.f,0.f,0.f,0.f,0.f,0.f};
  v8f acc[2][2] = {{z, z}, {z, z}};
  int T = K >> 6;
  stage(0, 0);
  for (int t = 0; t < T; t++) {
    int cur = t & 1;
    if (t + 1 < T) { stage((t + 1) << 6, cur ^ 1); async_wait6(); }
    else           { async_wait0(); }
    __syncthreads();            // tile t visible to all waves
    const u16t* cA = sA[cur];
    const u16t* cW = sW[cur];
    for (int ks = 0; ks < 64; ks += 32) {
      v16bf af0 = load_frag(cA, wm * 32,      ks, 64);
      v16bf af1 = load_frag(cA, wm * 32 + 16, ks, 64);
      v16bf bf0 = load_frag(cW, wn * 32,      ks, 64);
      v16bf bf1 = load_frag(cW, wn * 32 + 16, ks, 64);
      acc[0][0] = wmma_bf16(af0, bf0, acc[0][0]);
      acc[0][1] = wmma_bf16(af0, bf1, acc[0][1]);
      acc[1][0] = wmma_bf16(af1, bf0, acc[1][0]);
      acc[1][1] = wmma_bf16(af1, bf1, acc[1][1]);
    }
    __syncthreads();            // tile t consumed; its buffer may be re-staged
  }
  for (int mt = 0; mt < 2; mt++)
    for (int nt = 0; nt < 2; nt++)
      for (int i = 0; i < 8; i++) {
        int row = m0 + wm * 32 + mt * 16 + i + ((lane >> 4) << 3);
        int col = n0 + wn * 32 + nt * 16 + (lane & 15);
        float v = acc[mt][nt][i] + bias[col];
        if (relu) v = fmaxf(v, 0.f);
        size_t o = (size_t)row * N + col;
        if (resid) v += resid[o];
        if (outF) outF[o] = v;
        if (outB) outB[o] = f2b(v);
      }
}

// ---------------------------------------------------------------------------
// Flash attention (no mask). qkv bf16 [4096,1536]; vt bf16 [B*H][64][SEQ].
// out bf16 [4096,512].  block = 128 threads = 4 waves; 16 query rows/wave;
// 64-key chunks.  K and Vt chunks staged with async global->LDS copies.
// grid = B*H * (S/64)
// ---------------------------------------------------------------------------
__global__ __launch_bounds__(128) void attn_kernel(const u16t* __restrict__ qkv,
                                                   const u16t* __restrict__ vt,
                                                   u16t* __restrict__ out) {
  __shared__ u16t  sK[64 * 64];        // [key][d]
  __shared__ u16t  sV[64 * 64];        // [d][key]
  __shared__ float sS[4][16 * 64];     // per-wave scores
  __shared__ u16t  sP[4][16 * 64];     // per-wave probs (bf16, A layout image)
  __shared__ float sRow[4][16];        // per-wave row stats exchange
  int tid = threadIdx.x;
  int lane = tid & 31;
  int wid = tid >> 5;
  int blk = blockIdx.x;
  int qblk = blk & 31;        // 32 query blocks of 64
  int bh = blk >> 5;
  int b = bh >> 3;
  int h = bh & 7;
  int q0 = b * SEQ + qblk * 64 + wid * 16;
  unsigned sKoff = lds_offset(sK);
  unsigned sVoff = lds_offset(sV);
  const u16t* vtb = vt + (size_t)bh * DHEAD * SEQ;

  const u16t* qptr = qkv + (size_t)q0 * TD + h * DHEAD;   // q cols 0..511
  v16bf qf0 = load_frag(qptr, 0, 0, TD);
  v16bf qf1 = load_frag(qptr, 0, 32, TD);

  v8f z = {0.f,0.f,0.f,0.f,0.f,0.f,0.f,0.f};
  v8f oacc[4] = {z, z, z, z};
  float m_i = -1e30f, l_i = 0.f;
  int r = lane & 15, half = lane >> 4;
  float* ss = sS[wid];
  u16t*  sp = sP[wid];

  for (int kc = 0; kc < SEQ / 64; kc++) {
    __syncthreads();
    // async stage K chunk: 64 keys x 128B rows; and Vt chunk: 64 d-rows x 128B
    for (int rr = 0; rr < 4; rr++) {
      int lin = (tid + (rr << 7)) << 4;
      int row = lin >> 7;
      int db = lin & 127;
      async_copy_b128(sKoff + (unsigned)lin,
          (const char*)(qkv + (size_t)(b * SEQ + kc * 64 + row) * TD
                        + 512 + h * DHEAD) + db);
      async_copy_b128(sVoff + (unsigned)lin,
          (const char*)(vtb + (size_t)row * SEQ + kc * 64) + db);
    }
    async_wait0();
    __syncthreads();
    // scores: 16 queries x 64 keys
    v8f sacc[4] = {z, z, z, z};
    for (int nt = 0; nt < 4; nt++) {
      v16bf kb0 = load_frag(sK, nt * 16, 0, 64);
      v16bf kb1 = load_frag(sK, nt * 16, 32, 64);
      sacc[nt] = wmma_bf16(qf0, kb0, sacc[nt]);
      sacc[nt] = wmma_bf16(qf1, kb1, sacc[nt]);
    }
    for (int nt = 0; nt < 4; nt++)
      for (int i = 0; i < 8; i++)
        ss[(i + (half << 3)) * 64 + nt * 16 + (lane & 15)] = sacc[nt][i] * 0.125f;
    __syncthreads();
    // online softmax: lane handles row r, key-half `half`
    float mc = -1e30f;
    for (int j = 0; j < 32; j++) mc = fmaxf(mc, ss[r * 64 + half * 32 + j]);
    mc = fmaxf(mc, __shfl_xor(mc, 16, 32));
    float mnew = fmaxf(m_i, mc);
    float alpha = __expf(m_i - mnew);
    float psum = 0.f;
    for (int j = 0; j < 32; j++) {
      float p = __expf(ss[r * 64 + half * 32 + j] - mnew);
      psum += p;
      sp[r * 64 + half * 32 + j] = f2b(p);
    }
    psum += __shfl_xor(psum, 16, 32);
    l_i = l_i * alpha + psum;
    m_i = mnew;
    if (lane < 16) sRow[wid][lane] = alpha;
    __syncthreads();
    for (int nt = 0; nt < 4; nt++)
      for (int i = 0; i < 8; i++)
        oacc[nt][i] *= sRow[wid][i + (half << 3)];
    // P @ V
    v16bf pf0 = load_frag(sp, 0, 0, 64);
    v16bf pf1 = load_frag(sp, 0, 32, 64);
    for (int dt = 0; dt < 4; dt++) {
      v16bf vf0 = load_frag(sV, dt * 16, 0, 64);
      v16bf vf1 = load_frag(sV, dt * 16, 32, 64);
      oacc[dt] = wmma_bf16(pf0, vf0, oacc[dt]);
      oacc[dt] = wmma_bf16(pf1, vf1, oacc[dt]);
    }
  }
  if (lane < 16) sRow[wid][lane] = (l_i > 0.f) ? 1.f / l_i : 0.f;
  __syncthreads();
  for (int dt = 0; dt < 4; dt++)
    for (int i = 0; i < 8; i++) {
      int rr = i + (half << 3);
      int row = q0 + rr;
      int col = h * DHEAD + dt * 16 + (lane & 15);
      out[(size_t)row * D_MODEL + col] = f2b(oacc[dt][i] * sRow[wid][rr]);
    }
}

// ---------------------------------------------------------------------------
extern "C" void kernel_launch(void* const* d_in, const int* in_sizes, int n_in,
                              void* d_out, int out_size, void* d_ws, size_t ws_size,
                              hipStream_t stream) {
  (void)in_sizes; (void)n_in; (void)out_size; (void)ws_size;
  const int*   x      = (const int*)d_in[0];
  const float* tokemb = (const float*)d_in[1];
  const float* ln1_g  = (const float*)d_in[2];
  const float* ln1_b  = (const float*)d_in[3];
  const float* qkv_w  = (const float*)d_in[4];
  const float* qkv_b  = (const float*)d_in[5];
  const float* proj_w = (const float*)d_in[6];
  const float* proj_b = (const float*)d_in[7];
  const float* ln2_g  = (const float*)d_in[8];
  const float* ln2_b  = (const float*)d_in[9];
  const float* fc1_w  = (const float*)d_in[10];
  const float* fc1_b  = (const float*)d_in[11];
  const float* fc2_w  = (const float*)d_in[12];
  const float* fc2_b  = (const float*)d_in[13];
  const float* fln_g  = (const float*)d_in[14];
  const float* fln_b  = (const float*)d_in[15];
  const float* head_w = (const float*)d_in[16];
  const float* head_b = (const float*)d_in[17];

  char* ws = (char*)d_ws;
  size_t off = 0;
  auto alloc = [&](size_t bytes) -> void* {
    void* p = ws + off;
    off = (off + bytes + 255) & ~(size_t)255;
    return p;
  };
  float* h     = (float*)alloc((size_t)ROWS * D_MODEL * 4);
  u16t* lnbuf  = (u16t*)alloc((size_t)ROWS * D_MODEL * 2);
  u16t* qkvbuf = (u16t*)alloc((size_t)ROWS * TD * 2);
  u16t* attno  = (u16t*)alloc((size_t)ROWS * D_MODEL * 2);
  u16t* ff1    = (u16t*)alloc((size_t)ROWS * FFDIM * 2);
  u16t* vtbuf  = (u16t*)alloc((size_t)ROWS * D_MODEL * 2);
  u16t* wqkv   = (u16t*)alloc((size_t)NLAYER * D_MODEL * TD * 2);     // [N,K] per layer
  u16t* wproj  = (u16t*)alloc((size_t)NLAYER * D_MODEL * D_MODEL * 2);
  u16t* wfc1   = (u16t*)alloc((size_t)NLAYER * D_MODEL * FFDIM * 2);
  u16t* wfc2   = (u16t*)alloc((size_t)NLAYER * FFDIM * D_MODEL * 2);
  u16t* whead  = (u16t*)alloc((size_t)D_MODEL * VOCAB * 2);

  // transpose-convert all weights to bf16 [N,K]
  cvt_t_kernel<<<dim3(TD / 32, D_MODEL / 32, NLAYER), 256, 0, stream>>>(qkv_w, wqkv, D_MODEL, TD);
  cvt_t_kernel<<<dim3(D_MODEL / 32, D_MODEL / 32, NLAYER), 256, 0, stream>>>(proj_w, wproj, D_MODEL, D_MODEL);
  cvt_t_kernel<<<dim3(FFDIM / 32, D_MODEL / 32, NLAYER), 256, 0, stream>>>(fc1_w, wfc1, D_MODEL, FFDIM);
  cvt_t_kernel<<<dim3(D_MODEL / 32, FFDIM / 32, NLAYER), 256, 0, stream>>>(fc2_w, wfc2, FFDIM, D_MODEL);
  cvt_t_kernel<<<dim3(VOCAB / 32, D_MODEL / 32, 1), 256, 0, stream>>>(head_w, whead, D_MODEL, VOCAB);

  embed_kernel<<<ROWS, 512, 0, stream>>>(x, tokemb, h);

  for (int l = 0; l < NLAYER; l++) {
    ln_kernel<<<ROWS / 8, 256, 0, stream>>>(h, ln1_g + l * D_MODEL, ln1_b + l * D_MODEL, lnbuf);
    gemm_kernel<<<dim3(TD / 64, ROWS / 128), 256, 0, stream>>>(
        lnbuf, wqkv + (size_t)l * D_MODEL * TD, qkv_b + l * TD,
        nullptr, nullptr, qkvbuf, ROWS, TD, D_MODEL, 0);
    vtrans_kernel<<<dim3(SEQ / 32, DHEAD / 32, BATCH * NHEAD), 256, 0, stream>>>(qkvbuf, vtbuf);
    attn_kernel<<<BATCH * NHEAD * (SEQ / 64), 128, 0, stream>>>(qkvbuf, vtbuf, attno);
    gemm_kernel<<<dim3(D_MODEL / 64, ROWS / 128), 256, 0, stream>>>(
        attno, wproj + (size_t)l * D_MODEL * D_MODEL, proj_b + l * D_MODEL,
        h, h, nullptr, ROWS, D_MODEL, D_MODEL, 0);
    ln_kernel<<<ROWS / 8, 256, 0, stream>>>(h, ln2_g + l * D_MODEL, ln2_b + l * D_MODEL, lnbuf);
    gemm_kernel<<<dim3(FFDIM / 64, ROWS / 128), 256, 0, stream>>>(
        lnbuf, wfc1 + (size_t)l * D_MODEL * FFDIM, fc1_b + l * FFDIM,
        nullptr, nullptr, ff1, ROWS, FFDIM, D_MODEL, 1);
    gemm_kernel<<<dim3(D_MODEL / 64, ROWS / 128), 256, 0, stream>>>(
        ff1, wfc2 + (size_t)l * FFDIM * D_MODEL, fc2_b + l * D_MODEL,
        h, h, nullptr, ROWS, D_MODEL, FFDIM, 0);
  }
  ln_kernel<<<ROWS / 8, 256, 0, stream>>>(h, fln_g, fln_b, lnbuf);
  gemm_kernel<<<dim3(VOCAB / 64, ROWS / 128), 256, 0, stream>>>(
      lnbuf, whead, head_b, nullptr, (float*)d_out, nullptr, ROWS, VOCAB, D_MODEL, 0);
}
